// GNN_56650618634369
// MI455X (gfx1250) — compile-verified, matrix operand order checked by reading, not run on hardware
//
#include <hip/hip_runtime.h>
#include <hip/hip_bf16.h>

// ---------------------------------------------------------------------------
// CDNA5 (gfx1250) GIN forward. All GEMMs use v_wmma_f32_16x16x32_bf16.
// - Weights pre-packed into WMMA B-fragment layout (2x global_load_b128/frag)
// - B fragments register-resident, reused across 4 row-tiles (M=64/block)
// - LDS staging vectorized: 8 elems/thread -> 1x ds_store_b128
// - Streaming kernels vectorized to float4 (23.3 TB/s wants 128-bit accesses)
// h/aggr/hh (51.2MB each) are L2-resident (192MB L2).
// ---------------------------------------------------------------------------

typedef __attribute__((ext_vector_type(16))) __bf16 v16bf;
typedef __attribute__((ext_vector_type(8)))  float  v8f;

union BF16x16 { v16bf v; unsigned short s[16]; unsigned int u[8]; };

__device__ __forceinline__ unsigned short f2bf(float f) {
  unsigned int u = __builtin_bit_cast(unsigned int, f);
  u += 0x7FFFu + ((u >> 16) & 1u);          // round-to-nearest-even
  return (unsigned short)(u >> 16);
}

__device__ __forceinline__ unsigned int pack2bf(float a, float b) {
  return (unsigned int)f2bf(a) | ((unsigned int)f2bf(b) << 16);
}

// A fragment (16x32 bf16, MxK) from LDS row-major [16][ldk] bf16 at base A.
// Lane: m = lane&15, half = lane>>4. elem e -> K = (e/8)*16 + half*8 + (e%8).
// Contiguous dword groups -> compiler merges to ds_load_b128 x2.
__device__ __forceinline__ v16bf frag_a_lds(const unsigned short* A, int ldk, int k0) {
  int lane = threadIdx.x & 31;
  int m = lane & 15, half = lane >> 4;
  BF16x16 r;
  const unsigned short* p = A + m * ldk + k0 + half * 8;   // 4B aligned (ldk,k0 even)
#pragma unroll
  for (int g = 0; g < 2; ++g)
#pragma unroll
    for (int i = 0; i < 4; ++i)
      r.u[g * 4 + i] = *(const unsigned int*)(p + g * 16 + 2 * i);
  return r.v;
}

// B fragment from pre-packed global layout: frag -> [32 lanes][16 bf16] contig.
// Per lane: 32 contiguous bytes -> 2x global_load_b128.
__device__ __forceinline__ v16bf frag_b_packed(const unsigned short* Bp, int frag) {
  int lane = threadIdx.x & 31;
  const unsigned int* p = (const unsigned int*)(Bp + ((size_t)frag * 32 + lane) * 16);
  BF16x16 r;
#pragma unroll
  for (int i = 0; i < 8; ++i) r.u[i] = p[i];
  return r.v;
}

__device__ __forceinline__ v8f wmma_bf16(v16bf a, v16bf b, v8f c) {
  return __builtin_amdgcn_wmma_f32_16x16x32_bf16(false, a, false, b, (short)0, c, false, false);
}

// ---------------------------------------------------------------------------
// Pack fp32 weights [K][N] -> bf16 WMMA-B-fragment layout, zero K-padded.
// Fragment (tk, tn) stored at ((tk*(N/16) + tn)*32 + lane)*16 + e, where
// k = tk*32 + (e/8)*16 + (lane/16)*8 + (e%8),  n = tn*16 + (lane%16).
__global__ void k_pack_b(const float* __restrict__ in, unsigned short* __restrict__ out,
                         int K, int N, int Kpad) {
  int TN = N >> 4;
  int total = Kpad * N;
  for (int i = blockIdx.x * blockDim.x + threadIdx.x; i < total; i += gridDim.x * blockDim.x) {
    int e = i & 15;
    int lane = (i >> 4) & 31;
    int frag = i >> 9;
    int tn = frag % TN, tk = frag / TN;
    int k = tk * 32 + ((e >> 3) << 4) + (lane >> 4) * 8 + (e & 7);
    int n = tn * 16 + (lane & 15);
    out[i] = (k < K) ? f2bf(in[k * N + n]) : (unsigned short)0;
  }
}

// self-loop embedding: self_e = leaky(eb1) @ ew2 + eb2   (1x128 @ 128x128, fp32)
__global__ void k_selfe(const float* __restrict__ eb1, const float* __restrict__ ew2,
                        const float* __restrict__ eb2, float* __restrict__ se) {
  int c = threadIdx.x;
  if (c >= 128) return;
  float acc = eb2[c];
  for (int k = 0; k < 128; ++k) {
    float v = eb1[k];
    v = v > 0.f ? v : 0.1f * v;
    acc += v * ew2[k * 128 + c];
  }
  se[c] = acc;
}

// aggr = h + self_e  (float4 streaming: 32 col-groups of 4)
__global__ void k_init_aggr(const float4* __restrict__ h4, const float4* __restrict__ se4,
                            float4* __restrict__ aggr4, int total4) {
  for (int i = blockIdx.x * blockDim.x + threadIdx.x; i < total4; i += gridDim.x * blockDim.x) {
    float4 a = h4[i], s = se4[i & 31];
    float4 o; o.x = a.x + s.x; o.y = a.y + s.y; o.z = a.z + s.z; o.w = a.w + s.w;
    aggr4[i] = o;
  }
}

// ---------------------------------------------------------------------------
// Input MLP: h = leaky(x@iw1+ib1)@iw2+ib2    (x: [N,178] fp32), M=64/block
__global__ __launch_bounds__(256) void k_input(
    const float* __restrict__ x, const unsigned short* __restrict__ w1,
    const float* __restrict__ b1, const unsigned short* __restrict__ w2,
    const float* __restrict__ b2, float* __restrict__ h, int n_nodes)
{
  __shared__ __align__(16) unsigned short sh_x[64 * 192];
  __shared__ __align__(16) unsigned short sh_hid[64 * 128];
  int tid = threadIdx.x;
  int row0 = blockIdx.x * 64;

  for (int g = tid; g < (64 * 192) / 8; g += 256) {     // 8 elems / thread / step
    int r = g / 24, cg = (g - r * 24) * 8;              // 24 groups per 192-col row
    int node = row0 + r;
    const float* xr = x + (long)node * 178;
    uint4 q;
    unsigned int u[4];
#pragma unroll
    for (int j = 0; j < 4; ++j) {
      int c0 = cg + 2 * j;
      float v0 = (c0     < 178 && node < n_nodes) ? xr[c0]     : 0.f;
      float v1 = (c0 + 1 < 178 && node < n_nodes) ? xr[c0 + 1] : 0.f;
      u[j] = pack2bf(v0, v1);
    }
    q.x = u[0]; q.y = u[1]; q.z = u[2]; q.w = u[3];
    *(uint4*)(&sh_x[g * 8]) = q;                        // ds_store_b128
  }
  __syncthreads();

  int w = tid >> 5, lane = tid & 31, half = lane >> 4;
  int col = w * 16 + (lane & 15);

  {
    v16bf bf[6];                                        // K = 192 (178 padded), TN = 8
#pragma unroll
    for (int ks = 0; ks < 6; ++ks) bf[ks] = frag_b_packed(w1, ks * 8 + w);
    float bias1 = b1[col];
#pragma unroll
    for (int mt = 0; mt < 4; ++mt) {
      v8f c0 = {};
#pragma unroll
      for (int ks = 0; ks < 6; ++ks)
        c0 = wmma_bf16(frag_a_lds(sh_x + mt * 16 * 192, 192, ks * 32), bf[ks], c0);
#pragma unroll
      for (int r = 0; r < 8; ++r) {
        float v = c0[r] + bias1;
        v = v > 0.f ? v : 0.1f * v;                     // LeakyReLU(0.1)
        sh_hid[(mt * 16 + r + half * 8) * 128 + col] = f2bf(v);
      }
    }
  }
  __syncthreads();

  {
    v16bf bf[4];                                        // K = 128, TN = 8
#pragma unroll
    for (int ks = 0; ks < 4; ++ks) bf[ks] = frag_b_packed(w2, ks * 8 + w);
    float bias2 = b2[col];
#pragma unroll
    for (int mt = 0; mt < 4; ++mt) {
      v8f c1 = {};
#pragma unroll
      for (int ks = 0; ks < 4; ++ks)
        c1 = wmma_bf16(frag_a_lds(sh_hid + mt * 16 * 128, 128, ks * 32), bf[ks], c1);
#pragma unroll
      for (int r = 0; r < 8; ++r) {
        int node = row0 + mt * 16 + r + half * 8;
        if (node < n_nodes) h[(long)node * 128 + col] = c1[r] + bias2;
      }
    }
  }
}

// ---------------------------------------------------------------------------
// Edge pipeline: ee = leaky(ea@ew1+eb1)@ew2+eb2; atomic aggr[dst] += h[src]+ee
// M = 64 edges per block.
__global__ __launch_bounds__(256) void k_edge(
    const float* __restrict__ eattr, const int* __restrict__ eidx,
    const unsigned short* __restrict__ w1, const float* __restrict__ b1,
    const unsigned short* __restrict__ w2, const float* __restrict__ b2,
    const float* __restrict__ h, float* __restrict__ aggr,
    int n_edges)
{
  __shared__ __align__(16) unsigned short sh_e[64 * 32];
  __shared__ __align__(16) unsigned short sh_hid[64 * 128];
  __shared__ int sh_src[64], sh_dst[64];
  int tid = threadIdx.x;
  int e0 = blockIdx.x * 64;

  {                                                     // one 8-elem group / thread
    int r = tid >> 2, cg = (tid & 3) * 8;               // cols: 0..17 data, rest 0
    int e = e0 + r;
    const float* er = eattr + (long)e * 18;
    uint4 q;
    unsigned int u[4];
#pragma unroll
    for (int j = 0; j < 4; ++j) {
      int c0 = cg + 2 * j;
      float v0 = (c0     < 18 && e < n_edges) ? er[c0]     : 0.f;
      float v1 = (c0 + 1 < 18 && e < n_edges) ? er[c0 + 1] : 0.f;
      u[j] = pack2bf(v0, v1);
    }
    q.x = u[0]; q.y = u[1]; q.z = u[2]; q.w = u[3];
    *(uint4*)(&sh_e[tid * 8]) = q;
  }
  if (tid < 64) {
    int e = e0 + tid;
    sh_src[tid] = (e < n_edges) ? eidx[e] : 0;
    sh_dst[tid] = (e < n_edges) ? eidx[n_edges + e] : 0;
  }
  __syncthreads();

  int w = tid >> 5, lane = tid & 31, half = lane >> 4;
  int col = w * 16 + (lane & 15);

  {
    v16bf bf0 = frag_b_packed(w1, w);                   // K=32 (18 padded), TN=8
    float bias1 = b1[col];
#pragma unroll
    for (int mt = 0; mt < 4; ++mt) {
      v8f c0 = {};
      c0 = wmma_bf16(frag_a_lds(sh_e + mt * 16 * 32, 32, 0), bf0, c0);
#pragma unroll
      for (int r = 0; r < 8; ++r) {
        float v = c0[r] + bias1;
        v = v > 0.f ? v : 0.1f * v;
        sh_hid[(mt * 16 + r + half * 8) * 128 + col] = f2bf(v);
      }
    }
  }
  __syncthreads();

  {
    v16bf bf[4];                                        // K = 128, TN = 8
#pragma unroll
    for (int ks = 0; ks < 4; ++ks) bf[ks] = frag_b_packed(w2, ks * 8 + w);
    float bias2 = b2[col];
#pragma unroll
    for (int mt = 0; mt < 4; ++mt) {
      v8f c1 = {};
#pragma unroll
      for (int ks = 0; ks < 4; ++ks)
        c1 = wmma_bf16(frag_a_lds(sh_hid + mt * 16 * 128, 128, ks * 32), bf[ks], c1);
#pragma unroll
      for (int r = 0; r < 8; ++r) {
        int rr = mt * 16 + r + half * 8;
        int e = e0 + rr;
        if (e < n_edges) {
          float v = c1[r] + bias2 + h[(long)sh_src[rr] * 128 + col];  // gather (L2)
          atomicAdd(&aggr[(long)sh_dst[rr] * 128 + col], v);          // scatter-add (L2)
        }
      }
    }
  }
}

// ---------------------------------------------------------------------------
// GIN update MLP: hh = relu(aggr@mw1+mb1)@mw2+mb2   (128 -> 256 -> 128), M=64
__global__ __launch_bounds__(256) void k_gin(
    const float* __restrict__ aggr,
    const unsigned short* __restrict__ w1, const float* __restrict__ b1,
    const unsigned short* __restrict__ w2, const float* __restrict__ b2,
    float* __restrict__ hh, int n_nodes)
{
  __shared__ __align__(16) unsigned short sh_a[64 * 128];
  __shared__ __align__(16) unsigned short sh_hid[64 * 256];
  int tid = threadIdx.x;
  int row0 = blockIdx.x * 64;

  for (int g = tid; g < (64 * 128) / 8; g += 256) {     // 8 elems / thread / step
    int r = g >> 4, cg = (g & 15) * 8;
    int node = row0 + r;
    uint4 q;
    if (node < n_nodes) {
      const float4* p = (const float4*)(aggr + (long)node * 128 + cg);  // 32B aligned
      float4 a = p[0], b = p[1];
      q.x = pack2bf(a.x, a.y); q.y = pack2bf(a.z, a.w);
      q.z = pack2bf(b.x, b.y); q.w = pack2bf(b.z, b.w);
    } else {
      q.x = 0u; q.y = 0u; q.z = 0u; q.w = 0u;
    }
    *(uint4*)(&sh_a[g * 8]) = q;
  }
  __syncthreads();

  int w = tid >> 5, lane = tid & 31, half = lane >> 4;

#pragma unroll
  for (int t = 0; t < 2; ++t) {                         // hidden N=256 -> TN=16, 2 tiles/wave
    int tn = w * 2 + t;
    v16bf bf[4];
#pragma unroll
    for (int ks = 0; ks < 4; ++ks) bf[ks] = frag_b_packed(w1, ks * 16 + tn);
    int col = tn * 16 + (lane & 15);
    float bias = b1[col];
#pragma unroll
    for (int mt = 0; mt < 4; ++mt) {
      v8f c0 = {};
#pragma unroll
      for (int ks = 0; ks < 4; ++ks)
        c0 = wmma_bf16(frag_a_lds(sh_a + mt * 16 * 128, 128, ks * 32), bf[ks], c0);
#pragma unroll
      for (int r = 0; r < 8; ++r) {
        float v = fmaxf(c0[r] + bias, 0.f);             // ReLU
        sh_hid[(mt * 16 + r + half * 8) * 256 + col] = f2bf(v);
      }
    }
  }
  __syncthreads();

  {
    v16bf bf[8];                                        // K = 256, TN = 8
#pragma unroll
    for (int ks = 0; ks < 8; ++ks) bf[ks] = frag_b_packed(w2, ks * 8 + w);
    int col = w * 16 + (lane & 15);
    float bias2 = b2[col];
#pragma unroll
    for (int mt = 0; mt < 4; ++mt) {
      v8f c1 = {};
#pragma unroll
      for (int ks = 0; ks < 8; ++ks)
        c1 = wmma_bf16(frag_a_lds(sh_hid + mt * 16 * 256, 256, ks * 32), bf[ks], c1);
#pragma unroll
      for (int r = 0; r < 8; ++r) {
        int node = row0 + mt * 16 + r + half * 8;
        if (node < n_nodes) hh[(long)node * 128 + col] = c1[r] + bias2;
      }
    }
  }
}

// ---------------------------------------------------------------------------
// BatchNorm helpers
__global__ void k_zero(float* p, int n) {
  int i = blockIdx.x * blockDim.x + threadIdx.x;
  if (i < n) p[i] = 0.f;
}

// float4 per thread over a 4-column group; 8 atomics / thread at the end.
__global__ __launch_bounds__(128) void k_stats(const float4* __restrict__ hh4,
                                               float* __restrict__ sum,
                                               float* __restrict__ sumsq, int n) {
  int t = threadIdx.x;
  int cg = t & 31;                 // column group (4 cols)
  int rsub = t >> 5;               // 0..3
  float sx = 0.f, sy = 0.f, sz = 0.f, sw = 0.f;
  float qx = 0.f, qy = 0.f, qz = 0.f, qw = 0.f;
  for (int r = blockIdx.x * 4 + rsub; r < n; r += gridDim.x * 4) {
    float4 v = hh4[(long)r * 32 + cg];
    sx += v.x; sy += v.y; sz += v.z; sw += v.w;
    qx += v.x * v.x; qy += v.y * v.y; qz += v.z * v.z; qw += v.w * v.w;
  }
  int c = cg * 4;
  atomicAdd(&sum[c + 0], sx); atomicAdd(&sum[c + 1], sy);
  atomicAdd(&sum[c + 2], sz); atomicAdd(&sum[c + 3], sw);
  atomicAdd(&sumsq[c + 0], qx); atomicAdd(&sumsq[c + 1], qy);
  atomicAdd(&sumsq[c + 2], qz); atomicAdd(&sumsq[c + 3], qw);
}

__global__ void k_finstats(const float* __restrict__ sum, const float* __restrict__ sumsq,
                           const float* __restrict__ gamma, const float* __restrict__ beta,
                           float* __restrict__ scale, float* __restrict__ shift, float inv_n) {
  int c = threadIdx.x;
  if (c >= 128) return;
  float m  = sum[c] * inv_n;
  float var = fmaxf(sumsq[c] * inv_n - m * m, 0.f);     // biased variance
  float sc = gamma[c] * rsqrtf(var + 1e-5f);
  scale[c] = sc;
  shift[c] = beta[c] - m * sc;
}

__global__ void k_bn(const float4* __restrict__ in4, const float4* __restrict__ scale4,
                     const float4* __restrict__ shift4, float4* __restrict__ out4,
                     int total4, int relu) {
  for (int i = blockIdx.x * blockDim.x + threadIdx.x; i < total4; i += gridDim.x * blockDim.x) {
    float4 v = in4[i];
    float4 sc = scale4[i & 31], sh = shift4[i & 31];
    float4 o;
    o.x = v.x * sc.x + sh.x; o.y = v.y * sc.y + sh.y;
    o.z = v.z * sc.z + sh.z; o.w = v.w * sc.w + sh.w;
    if (relu) {
      o.x = fmaxf(o.x, 0.f); o.y = fmaxf(o.y, 0.f);
      o.z = fmaxf(o.z, 0.f); o.w = fmaxf(o.w, 0.f);
    }
    out4[i] = o;
  }
}

// ---------------------------------------------------------------------------
extern "C" void kernel_launch(void* const* d_in, const int* in_sizes, int n_in,
                              void* d_out, int out_size, void* d_ws, size_t ws_size,
                              hipStream_t stream) {
  const float* x     = (const float*)d_in[0];
  const float* eattr = (const float*)d_in[1];
  const float* iw1   = (const float*)d_in[2];
  const float* ib1   = (const float*)d_in[3];
  const float* iw2   = (const float*)d_in[4];
  const float* ib2   = (const float*)d_in[5];
  const float* ew1   = (const float*)d_in[6];
  const float* eb1   = (const float*)d_in[7];
  const float* ew2   = (const float*)d_in[8];
  const float* eb2   = (const float*)d_in[9];
  const float* mw1   = (const float*)d_in[10];
  const float* mb1   = (const float*)d_in[11];
  const float* mw2   = (const float*)d_in[12];
  const float* mb2   = (const float*)d_in[13];
  const float* gamma = (const float*)d_in[14];
  const float* beta  = (const float*)d_in[15];
  const int*   eidx  = (const int*)d_in[16];

  const int n_nodes = in_sizes[0] / 178;
  const int n_edges = in_sizes[1] / 18;
  const int total   = n_nodes * 128;
  const int NL = 5;

  // ---- workspace carve-out (~104 MB) ----
  char* ws = (char*)d_ws;
  size_t off = 0;
  auto carve = [&](size_t bytes) -> void* {
    void* p = ws + off;
    off = (off + bytes + 255) & ~(size_t)255;
    return p;
  };
  float* buf_h = (float*)carve((size_t)total * 4);      // h / hh (overwritten per layer)
  float* buf_a = (float*)carve((size_t)total * 4);      // aggr
  unsigned short* iw1b = (unsigned short*)carve((size_t)192 * 128 * 2);
  unsigned short* iw2b = (unsigned short*)carve((size_t)128 * 128 * 2);
  unsigned short* ew1b = (unsigned short*)carve((size_t)NL * 32 * 128 * 2);
  unsigned short* ew2b = (unsigned short*)carve((size_t)NL * 128 * 128 * 2);
  unsigned short* mw1b = (unsigned short*)carve((size_t)NL * 128 * 256 * 2);
  unsigned short* mw2b = (unsigned short*)carve((size_t)NL * 256 * 128 * 2);
  float* se    = (float*)carve(128 * 4);
  float* stats = (float*)carve(256 * 4);                // [0..127]=sum, [128..255]=sumsq
  float* scale = (float*)carve(128 * 4);
  float* shift = (float*)carve(128 * 4);

  // ---- one-time fp32 -> packed bf16 weight fragments (per matrix) ----
  k_pack_b<<<96, 256, 0, stream>>>(iw1, iw1b, 178, 128, 192);
  k_pack_b<<<64, 256, 0, stream>>>(iw2, iw2b, 128, 128, 128);
  for (int l = 0; l < NL; ++l) {
    k_pack_b<<<16, 256, 0, stream>>>(ew1 + (size_t)l * 18 * 128,
                                     ew1b + (size_t)l * 32 * 128, 18, 128, 32);
    k_pack_b<<<64, 256, 0, stream>>>(ew2 + (size_t)l * 128 * 128,
                                     ew2b + (size_t)l * 128 * 128, 128, 128, 128);
    k_pack_b<<<128, 256, 0, stream>>>(mw1 + (size_t)l * 128 * 256,
                                      mw1b + (size_t)l * 128 * 256, 128, 256, 128);
    k_pack_b<<<128, 256, 0, stream>>>(mw2 + (size_t)l * 256 * 128,
                                      mw2b + (size_t)l * 256 * 128, 256, 128, 256);
  }

  // ---- input MLP ----
  k_input<<<(n_nodes + 63) / 64, 256, 0, stream>>>(x, iw1b, ib1, iw2b, ib2, buf_h, n_nodes);

  // ---- layers ----
  for (int l = 0; l < NL; ++l) {
    k_selfe<<<1, 128, 0, stream>>>(eb1 + (size_t)l * 128, ew2 + (size_t)l * 128 * 128,
                                   eb2 + (size_t)l * 128, se);
    k_init_aggr<<<1024, 256, 0, stream>>>((const float4*)buf_h, (const float4*)se,
                                          (float4*)buf_a, total / 4);
    k_edge<<<(n_edges + 63) / 64, 256, 0, stream>>>(
        eattr, eidx,
        ew1b + (size_t)l * 32 * 128, eb1 + (size_t)l * 128,
        ew2b + (size_t)l * 128 * 128, eb2 + (size_t)l * 128,
        buf_h, buf_a, n_edges);
    k_gin<<<(n_nodes + 63) / 64, 256, 0, stream>>>(
        buf_a,
        mw1b + (size_t)l * 128 * 256, mb1 + (size_t)l * 256,
        mw2b + (size_t)l * 256 * 128, mb2 + (size_t)l * 128,
        buf_h, n_nodes);
    k_zero<<<1, 256, 0, stream>>>(stats, 256);
    k_stats<<<512, 128, 0, stream>>>((const float4*)buf_h, stats, stats + 128, n_nodes);
    k_finstats<<<1, 128, 0, stream>>>(stats, stats + 128,
                                      gamma + (size_t)l * 128, beta + (size_t)l * 128,
                                      scale, shift, 1.0f / (float)n_nodes);
    float* dst = (l == NL - 1) ? (float*)d_out : buf_h;
    k_bn<<<1024, 256, 0, stream>>>((const float4*)buf_h, (const float4*)scale,
                                   (const float4*)shift, (float4*)dst, total / 4,
                                   (l < NL - 1) ? 1 : 0);
  }
}
